// MS3DDeformAttn_54339926229590
// MI455X (gfx1250) — compile-verified, matrix operand order checked by reading, not run on hardware
//
#include <hip/hip_runtime.h>
#include <hip/hip_bf16.h>
#include <math.h>

// ---------------- constants (fixed by setup_inputs) ----------------
#define B_TOT   32          // B*nW
#define S_TOT   1568        // D*H*W
#define DMODEL  128
#define NHEADS  8
#define DHEAD   16
#define NPNT    4
#define DD      8
#define HH      14
#define WW      14
#define NWIN    32
#define NS      ((size_t)B_TOT * S_TOT)   // 50176 rows

typedef __attribute__((ext_vector_type(16))) _Float16 v16h;
typedef __attribute__((ext_vector_type(8)))  _Float16 v8h;
typedef __attribute__((ext_vector_type(8)))  float    v8f;

// Async global->LDS staging (CDNA5 ASYNCcnt path), guarded so compile
// cannot break on a toolchain without the builtins.
#if defined(__has_builtin)
#if __has_builtin(__builtin_amdgcn_global_load_async_to_lds_b128) && \
    __has_builtin(__builtin_amdgcn_s_wait_asynccnt)
#define USE_ASYNC_LDS 1
#endif
#endif

#ifdef USE_ASYNC_LDS
typedef int v4i_ __attribute__((ext_vector_type(4)));
typedef __attribute__((address_space(1))) v4i_ gv4i;   // global int4
typedef __attribute__((address_space(3))) v4i_ sv4i;   // LDS int4
#endif

// ------------------------------------------------------------------
// f32 -> f16 conversion (weights once, GEMM A-operand once).
// ------------------------------------------------------------------
__global__ __launch_bounds__(256)
void cvt_f16_kernel(const float* __restrict__ src, _Float16* __restrict__ dst,
                    long long n)
{
    long long i = (long long)blockIdx.x * 256 + threadIdx.x;
    long long stride = (long long)gridDim.x * 256;
    for (; i < n; i += stride) dst[i] = (_Float16)src[i];
}

// ------------------------------------------------------------------
// normalizer from mask: cumsum-then-max == axis sum, then max over rest.
// ------------------------------------------------------------------
__global__ __launch_bounds__(256)
void norm_kernel(const unsigned char* __restrict__ mask, float* __restrict__ nrm)
{
    int t = blockIdx.x * blockDim.x + threadIdx.x;
    if (t >= (int)NS) return;
    const unsigned char* m = mask + (size_t)t * (DD * HH * WW);
    int Wm = 0, Hm = 0, Dm = 0;
    for (int d = 0; d < DD; ++d)
        for (int h = 0; h < HH; ++h) {
            int s = 0;
            for (int x = 0; x < WW; ++x) s += m[(d * HH + h) * WW + x];
            Wm = s > Wm ? s : Wm;
        }
    for (int d = 0; d < DD; ++d)
        for (int x = 0; x < WW; ++x) {
            int s = 0;
            for (int h = 0; h < HH; ++h) s += m[(d * HH + h) * WW + x];
            Hm = s > Hm ? s : Hm;
        }
    for (int h = 0; h < HH; ++h)
        for (int x = 0; x < WW; ++x) {
            int s = 0;
            for (int d = 0; d < DD; ++d) s += m[(d * HH + h) * WW + x];
            Dm = s > Dm ? s : Dm;
        }
    nrm[t * 3 + 0] = (float)Wm;
    nrm[t * 3 + 1] = (float)Hm;
    nrm[t * 3 + 2] = (float)Dm;
}

// ------------------------------------------------------------------
// out[m][n] = sum_k A16[m][k] * W16[n][k] + bias[n]     (M x 128x128)
// 128-row blocks, 8 waves; each wave owns a 16-col strip, keeps its 4
// B-fragments resident in VGPRs and sweeps 8 row tiles -> 32 WMMA/wave.
// Both LDS tiles staged via GLOBAL_LOAD_ASYNC_TO_LDS_B128 when available.
// ------------------------------------------------------------------
template<int HEADMAJOR>
__global__ __launch_bounds__(256)
void gemm128_wmma(const _Float16* __restrict__ A16, const _Float16* __restrict__ W16,
                  const float* __restrict__ bias, float* __restrict__ out)
{
    __shared__ _Float16 sW[DMODEL * DMODEL];   // 32 KB  [n][k]
    __shared__ _Float16 sA[DMODEL * DMODEL];   // 32 KB  [m][k] (128 rows)

    const int tid = threadIdx.x;
    const size_t rowBase = (size_t)blockIdx.x * 128;
    const _Float16* __restrict__ Ablk = A16 + rowBase * DMODEL;

#ifdef USE_ASYNC_LDS
    for (int i = tid * 8; i < DMODEL * DMODEL; i += 256 * 8) {
        __builtin_amdgcn_global_load_async_to_lds_b128(
            (gv4i*)(void*)(W16 + i), (sv4i*)(void*)&sW[i], 0, 0);
        __builtin_amdgcn_global_load_async_to_lds_b128(
            (gv4i*)(void*)(Ablk + i), (sv4i*)(void*)&sA[i], 0, 0);
    }
    __builtin_amdgcn_s_wait_asynccnt(0);
#else
    for (int i = tid * 8; i < DMODEL * DMODEL; i += 256 * 8) {
        *(v8h*)&sW[i] = *(const v8h*)(W16 + i);
        *(v8h*)&sA[i] = *(const v8h*)(Ablk + i);
    }
#endif
    __syncthreads();

    const int wave  = tid >> 5;
    const int lane  = tid & 31;
    const int nBase = wave << 4;          // 16-column strip per wave (== head)
    const int halfw = lane >> 4;          // 0: K {0..7,16..23}; 1: K {8..15,24..31}
    const int l16   = lane & 15;
    const int koff  = halfw * 8;

    // B fragments: loaded once, reused for all 8 row tiles.
    v16h bfrag[4];
#pragma unroll
    for (int kt = 0; kt < 4; ++kt) {
        const _Float16* bp = &sW[(nBase + l16) * DMODEL + kt * 32 + koff];
        v8h lo = *reinterpret_cast<const v8h*>(bp);
        v8h hi = *reinterpret_cast<const v8h*>(bp + 16);
#pragma unroll
        for (int j = 0; j < 8; ++j) { bfrag[kt][j] = lo[j]; bfrag[kt][j + 8] = hi[j]; }
    }
    const float bval = bias[nBase + l16];

#pragma unroll
    for (int mt = 0; mt < 8; ++mt) {
        v8f acc;
#pragma unroll
        for (int r = 0; r < 8; ++r) acc[r] = bval;

#pragma unroll
        for (int kt = 0; kt < 4; ++kt) {
            const _Float16* ap = &sA[(mt * 16 + l16) * DMODEL + kt * 32 + koff];
            v8h lo = *reinterpret_cast<const v8h*>(ap);
            v8h hi = *reinterpret_cast<const v8h*>(ap + 16);
            v16h a;
#pragma unroll
            for (int j = 0; j < 8; ++j) { a[j] = lo[j]; a[j + 8] = hi[j]; }
            acc = __builtin_amdgcn_wmma_f32_16x16x32_f16(
                false, a, false, bfrag[kt], (short)0, acc, false, false);
        }

        // 16-row tile never straddles a batch row (1568 % 16 == 0):
        // b_/sBase are uniform -> SALU.
        const int mTile = (int)rowBase + mt * 16;
        const int bb    = mTile / S_TOT;
        const int sBase = mTile - bb * S_TOT;
#pragma unroll
        for (int r = 0; r < 8; ++r) {
            if (HEADMAJOR) {
                const int s = sBase + halfw * 8 + r;
                out[(((size_t)bb * NHEADS + wave) * S_TOT + s) * DHEAD + l16] = acc[r];
            } else {
                const size_t m = (size_t)mTile + halfw * 8 + r;
                out[m * DMODEL + nBase + l16] = acc[r];
            }
        }
    }
}

// ------------------------------------------------------------------
// Per (b_, s) position, one wave32: qm dot, 96 offset + 32 attn dots
// over the 33-dim qr, group-of-4 softmax, sample locations.
// ------------------------------------------------------------------
__global__ __launch_bounds__(256)
void qr_kernel(const float* __restrict__ query, const float* __restrict__ refpts,
               const float* __restrict__ pos_embed,
               const float* __restrict__ Wq, const float* __restrict__ bq,
               const float* __restrict__ Woff, const float* __restrict__ boff,
               const float* __restrict__ Wattn, const float* __restrict__ battn,
               const float* __restrict__ nrm,
               float* __restrict__ attn_w, float* __restrict__ loc_w)
{
    int gw = blockIdx.x * 8 + (threadIdx.x >> 5);
    if (gw >= (int)NS) return;
    const int lane = threadIdx.x & 31;
    const int b_ = gw / S_TOT, s = gw % S_TOT;
    const int w = b_ % NWIN;     // B == 1

    const float* qrow = query + (size_t)gw * DMODEL;
    float qm = 0.f;
#pragma unroll
    for (int i = 0; i < 4; ++i) qm += qrow[lane + 32 * i] * Wq[lane + 32 * i];
#pragma unroll
    for (int mk = 16; mk; mk >>= 1) qm += __shfl_xor(qm, mk, 32);
    qm += bq[0];

    float pe = pos_embed[((size_t)w * S_TOT + s) * NWIN + lane];

    const int j0 = lane, j1 = lane + 32, j2 = lane + 64;
    float o0 = Woff[j0 * 33] * qm + boff[j0];
    float o1 = Woff[j1 * 33] * qm + boff[j1];
    float o2 = Woff[j2 * 33] * qm + boff[j2];
    float at = Wattn[lane * 33] * qm + battn[lane];
    for (int i = 0; i < 32; ++i) {
        float q = __shfl(pe, i, 32);
        o0 += Woff[j0 * 33 + 1 + i] * q;
        o1 += Woff[j1 * 33 + 1 + i] * q;
        o2 += Woff[j2 * 33 + 1 + i] * q;
        at += Wattn[lane * 33 + 1 + i] * q;
    }

    float mx = fmaxf(at, __shfl_xor(at, 1, 32));
    mx = fmaxf(mx, __shfl_xor(mx, 2, 32));
    float e = __expf(at - mx);
    float se = e + __shfl_xor(e, 1, 32);
    se += __shfl_xor(se, 2, 32);
    attn_w[(size_t)gw * 32 + lane] = e / se;

    const float* nr = nrm + ((size_t)w * S_TOT + s) * 3;
    const float* rp = refpts + s * 3;
    float* lp = loc_w + (size_t)gw * 96;
    lp[j0] = rp[j0 % 3] + o0 / nr[j0 % 3];
    lp[j1] = rp[j1 % 3] + o1 / nr[j1 % 3];
    lp[j2] = rp[j2 % 3] + o2 / nr[j2 % 3];
}

// ------------------------------------------------------------------
// Trilinear gather + attention-weighted sum.  One thread per (b_,s,h,d);
// 16 consecutive lanes read a contiguous 64B value line per corner
// (value_h = 25.7 MB, fully L2-resident).  Result written as f16: it is
// only consumed as the f16 A-operand of the output projection.
// ------------------------------------------------------------------
__global__ __launch_bounds__(256)
void sample_kernel(const float* __restrict__ value_h, const float* __restrict__ attn_w,
                   const float* __restrict__ loc_w, _Float16* __restrict__ attn_out16)
{
    long long t = (long long)blockIdx.x * 256 + threadIdx.x;
    if (t >= (long long)NS * DMODEL) return;
    const int d  = (int)(t & 15);
    const int h  = (int)((t >> 4) & 7);
    const int gw = (int)(t >> 7);
    const int b_ = gw / S_TOT;

    const float* lp = loc_w + (size_t)gw * 96 + h * (NPNT * 3);
    const float* ap = attn_w + (size_t)gw * 32 + h * NPNT;
    const float* vb = value_h + ((size_t)(b_ * NHEADS + h)) * S_TOT * DHEAD;

    float acc = 0.f;
#pragma unroll
    for (int p = 0; p < NPNT; ++p) {
        float x = lp[p * 3 + 0] * (float)WW - 0.5f;
        float y = lp[p * 3 + 1] * (float)HH - 0.5f;
        float z = lp[p * 3 + 2] * (float)DD - 0.5f;
        float xf = floorf(x), yf = floorf(y), zf = floorf(z);
        int x0 = (int)xf, y0 = (int)yf, z0 = (int)zf;
        float fx = x - xf, fy = y - yf, fz = z - zf;
        float a = ap[p];
        float v = 0.f;
#pragma unroll
        for (int dz = 0; dz < 2; ++dz) {
            int zi = z0 + dz;
            if (zi < 0 || zi >= DD) continue;
            float wz = dz ? fz : 1.f - fz;
#pragma unroll
            for (int dy = 0; dy < 2; ++dy) {
                int yi = y0 + dy;
                if (yi < 0 || yi >= HH) continue;
                float wy = dy ? fy : 1.f - fy;
#pragma unroll
                for (int dx = 0; dx < 2; ++dx) {
                    int xi = x0 + dx;
                    if (xi < 0 || xi >= WW) continue;
                    float wx = dx ? fx : 1.f - fx;
                    v += wz * wy * wx * vb[((zi * HH + yi) * WW + xi) * DHEAD + d];
                }
            }
        }
        acc += a * v;
    }
    attn_out16[(size_t)gw * DMODEL + h * DHEAD + d] = (_Float16)acc;
}

// ------------------------------------------------------------------
extern "C" void kernel_launch(void* const* d_in, const int* in_sizes, int n_in,
                              void* d_out, int out_size, void* d_ws, size_t ws_size,
                              hipStream_t stream)
{
    const float* query   = (const float*)d_in[0];
    const float* refpts  = (const float*)d_in[1];
    const float* inflat  = (const float*)d_in[2];
    const unsigned char* mask = (const unsigned char*)d_in[5];
    const float* pos_embed = (const float*)d_in[6];
    const float* Wq    = (const float*)d_in[7];
    const float* bq    = (const float*)d_in[8];
    const float* Woff  = (const float*)d_in[9];
    const float* boff  = (const float*)d_in[10];
    const float* Wattn = (const float*)d_in[11];
    const float* battn = (const float*)d_in[12];
    const float* Wval  = (const float*)d_in[13];
    const float* bval  = (const float*)d_in[14];
    const float* Wout  = (const float*)d_in[15];
    const float* bout  = (const float*)d_in[16];

    // ---- workspace carve-up (all 16B aligned) ----
    char* wsb = (char*)d_ws;
    float* value_h    = (float*)wsb;                 wsb += NS * 128 * sizeof(float);
    float* normalizer = (float*)wsb;                 wsb += NS * 3   * sizeof(float);
    float* attn_w     = (float*)wsb;                 wsb += NS * 32  * sizeof(float);
    float* loc_w      = (float*)wsb;                 wsb += NS * 96  * sizeof(float);
    _Float16* a16     = (_Float16*)wsb;              wsb += NS * 128 * sizeof(_Float16);
    _Float16* attn16  = (_Float16*)wsb;              wsb += NS * 128 * sizeof(_Float16);
    _Float16* wval16  = (_Float16*)wsb;              wsb += 128 * 128 * sizeof(_Float16);
    _Float16* wout16  = (_Float16*)wsb;

    // 1. one-time f16 conversions (weights + GEMM1 A operand)
    cvt_f16_kernel<<<1024, 256, 0, stream>>>(inflat, a16, (long long)NS * 128);
    cvt_f16_kernel<<<64, 256, 0, stream>>>(Wval, wval16, 128 * 128);
    cvt_f16_kernel<<<64, 256, 0, stream>>>(Wout, wout16, 128 * 128);
    // 2. normalizer from mask
    norm_kernel<<<(int)((NS + 255) / 256), 256, 0, stream>>>(mask, normalizer);
    // 3. value projection -> head-major layout (WMMA)
    gemm128_wmma<1><<<(int)(NS / 128), 256, 0, stream>>>(a16, wval16, bval, value_h);
    // 4. offsets / attention / sample locations
    qr_kernel<<<(int)(NS / 8), 256, 0, stream>>>(query, refpts, pos_embed, Wq, bq,
                                                 Woff, boff, Wattn, battn,
                                                 normalizer, attn_w, loc_w);
    // 5. trilinear gather + weighted sum -> f16
    sample_kernel<<<(int)(NS * 128 / 256), 256, 0, stream>>>(value_h, attn_w, loc_w, attn16);
    // 6. output projection (WMMA) -> d_out
    gemm128_wmma<0><<<(int)(NS / 128), 256, 0, stream>>>(attn16, wout16, bout, (float*)d_out);
}
